// Brain_84301618086488
// MI455X (gfx1250) — matrix operand order repaired and checked by reading, not compiled, not run on hardware
//
// CDNA5 / gfx1250 implementation of the 5x5-grid "Brain" model.
//
// Strategy: everything is lowered to f16-in / f32-accum WMMA GEMMs
// (v_wmma_f32_16x16x32_f16, wave32). Convs use explicit im2col with K padded
// to a multiple of 32; BN+bias+ReLU fused into the GEMM epilogue. The 25
// per-node MLPs run as z-batched GEMMs (tanh epilogue, then a 600->10 logits
// GEMM with K padded to 608 / N padded to 16), followed by f32 softmax,
// neighbor gather, the second MLP pass, and the node-mean reduction.
//
// GEMM inner loop is double-buffered: the A tile for step k+1 is copied
// global->LDS with GLOBAL_LOAD_ASYNC_TO_LDS_B128 (ASYNCcnt) while step k's
// WMMAs execute; the (transposed-scatter) B tile load is issued before the
// compute block and scattered to LDS after it, so both copies hide under the
// matrix work.  s_wait_asynccnt 0 + workgroup barrier close each step.
//
// Workspace requirement: ~1.8 GB in d_ws (bump allocated, lifetime reuse).

#include <hip/hip_runtime.h>
#include <math.h>

typedef _Float16 h8  __attribute__((ext_vector_type(8)));
typedef _Float16 h16 __attribute__((ext_vector_type(16)));
typedef float    f8  __attribute__((ext_vector_type(8)));

#define TB 256            // threads per block (8 waves)
static const int TBM  = 128;   // block tile M
static const int TBN  = 64;    // block tile N
static const int TBK  = 32;    // K step == one WMMA K
static const int LDSK = 40;    // padded LDS K-stride (halves): conflict-free b128 loads

static const int GEMM_BUFH      = (TBM + TBN) * LDSK;   // halves per LDS buffer (7680)
static const int GEMM_LDS_BYTES = 2 * GEMM_BUFH * 2;    // double buffered (30720 B)

enum { EPI_RAW = 0, EPI_BNRELU = 1, EPI_TANH = 2, EPI_BIASF32 = 3 };

// ---- CDNA5 async global->LDS copy (per-lane 16B), tracked by ASYNCcnt ----
__device__ inline void async_load_b128(unsigned lds_byte_off, const void* gsrc) {
  asm volatile("global_load_async_to_lds_b128 %0, %1, off"
               :: "v"(lds_byte_off), "v"(gsrc)
               : "memory");
}
__device__ inline void wait_asynccnt0() {
  asm volatile("s_wait_asynccnt 0x0" ::: "memory");
}

// ---------------------------------------------------------------------------
// Generic f16 WMMA GEMM:  C[z] = A[z](MxK, row-major, lda) * B[z](KxN, ldb)
// Requirements: K % 32 == 0, N % 8 == 0 (true for every call site).
// For n in [N, nPad): stores zero (used to zero K-padding columns of h).
// ---------------------------------------------------------------------------
template <int EPI>
__global__ __launch_bounds__(TB) void gemm_wmma(
    const _Float16* __restrict__ A, int lda, long aStrideZ,
    const _Float16* __restrict__ Bm, int ldb, long bStrideZ,
    void* __restrict__ Cv, int ldc, long cStrideZ,
    int M, int N, int K, int nPad,
    const float* __restrict__ p0, const float* __restrict__ p1, int pStrideZ)
{
  extern __shared__ _Float16 lds[];  // [2][GEMM_BUFH]; sA at +0, sBt at +TBM*LDSK

  const int tid  = threadIdx.x;
  const int lane = tid & 31;
  const int wave = tid >> 5;
  const int z    = blockIdx.z;

  const _Float16* Az = A + (long)z * aStrideZ;
  const _Float16* Bz = Bm + (long)z * bStrideZ;

  const int mBase = blockIdx.y * TBM;
  const int nBase = blockIdx.x * TBN;
  const int m0 = (wave >> 1) * 32;   // 4 waves along M
  const int n0 = (wave & 1) * 32;    // 2 waves along N
  const int laneM = lane & 15;
  const int laneH = lane >> 4;

  f8 acc[2][2] = {};

  // global->LDS staging assignments
  const int arow = tid >> 1;                     // 0..127
  const int aseg = (tid & 1) * 16;               // 0 / 16 (halves)
  const int bk   = tid >> 3;                     // 0..31
  const int bcol = (tid & 7) * 8;                // 0..56

  const unsigned aLdsByte = (unsigned)((arow * LDSK + aseg) * 2);
  const _Float16* aGlob   = Az + (long)(mBase + arow) * lda + aseg;  // + kBase
  const bool aValid = (mBase + arow) < M;
  const int  gn     = nBase + bcol;
  const bool bValid = gn < N;                    // N % 8 == 0 -> whole-vector guard

  // async A tile stage (two 16B chunks per thread)
  auto stageA = [&](int kBase, int buf) {
    if (aValid) {
      const _Float16* src = aGlob + kBase;
      const unsigned off = (unsigned)(buf * GEMM_BUFH * 2) + aLdsByte;
      async_load_b128(off, src);
      async_load_b128(off + 16, src + 8);
    }
  };
  auto loadB = [&](int kBase) -> h8 {
    h8 v = {};
    if (bValid) v = *(const h8*)(Bz + (long)(kBase + bk) * ldb + gn);
    return v;
  };
  auto storeB = [&](h8 v, int buf) {  // transpose scatter into [n][k]
    _Float16* sBt = lds + buf * GEMM_BUFH + TBM * LDSK;
#pragma unroll
    for (int i = 0; i < 8; ++i) sBt[(bcol + i) * LDSK + bk] = v[i];
  };
  auto compute = [&](int buf) {
    const _Float16* sA  = lds + buf * GEMM_BUFH;
    const _Float16* sBt = sA + TBM * LDSK;
    // Fragments per ISA layout tables.
    // A 16x32: lane = M%16 (+half), elems 0..7 -> K=half*8+0..7, 8..15 -> K=16+half*8+0..7
    // B 32x16 (from transposed LDS): elems 0..7 -> K=half*8+0..7, 8..15 -> K=16+half*8+0..7 at column N=lane%16
    h16 afrag[2], bfrag[2];
#pragma unroll
    for (int am = 0; am < 2; ++am) {
      const _Float16* p = sA + (m0 + am * 16 + laneM) * LDSK + laneH * 8;
      h8 lo = *(const h8*)(p);
      h8 hi = *(const h8*)(p + 16);
      afrag[am] = __builtin_shufflevector(lo, hi, 0,1,2,3,4,5,6,7,8,9,10,11,12,13,14,15);
    }
#pragma unroll
    for (int bt = 0; bt < 2; ++bt) {
      const _Float16* p = sBt + (n0 + bt * 16 + laneM) * LDSK + laneH * 8;
      h8 lo = *(const h8*)(p);
      h8 hi = *(const h8*)(p + 16);
      bfrag[bt] = __builtin_shufflevector(lo, hi, 0,1,2,3,4,5,6,7,8,9,10,11,12,13,14,15);
    }
#pragma unroll
    for (int am = 0; am < 2; ++am)
#pragma unroll
      for (int bt = 0; bt < 2; ++bt)
        acc[am][bt] = __builtin_amdgcn_wmma_f32_16x16x32_f16(
            false, afrag[am], false, bfrag[bt], (short)0, acc[am][bt], false, false);
  };

  // ---- double-buffered main loop ----
  const int nk = K / TBK;
  stageA(0, 0);
  storeB(loadB(0), 0);
  wait_asynccnt0();
  __syncthreads();
  for (int kt = 0; kt < nk; ++kt) {
    const int cur = kt & 1;
    const bool more = (kt + 1) < nk;
    h8 breg = {};
    if (more) {
      stageA((kt + 1) * TBK, cur ^ 1);   // async, lands during compute
      breg = loadB((kt + 1) * TBK);      // global load in flight during compute
    }
    compute(cur);
    if (more) {
      storeB(breg, cur ^ 1);
      wait_asynccnt0();
    }
    __syncthreads();
  }

  // Epilogue. C/D layout: vgpr r, lanes0-15 -> M=r, lanes16-31 -> M=8+r; N=lane%16.
  _Float16* outH = (_Float16*)Cv + (long)z * cStrideZ;
  float*    outF = (float*)Cv + (long)z * cStrideZ;
#pragma unroll
  for (int am = 0; am < 2; ++am) {
#pragma unroll
    for (int bt = 0; bt < 2; ++bt) {
#pragma unroll
      for (int r = 0; r < 8; ++r) {
        const int row = mBase + m0 + am * 16 + r + laneH * 8;
        const int col = nBase + n0 + bt * 16 + laneM;
        if (row >= M) continue;
        if (col < N) {
          float v = acc[am][bt][r];
          if (EPI == EPI_BNRELU) {
            const float* al = p0 + (long)z * pStrideZ;
            const float* be = p1 + (long)z * pStrideZ;
            v = v * al[col] + be[col];
            v = v > 0.f ? v : 0.f;
            outH[(long)row * ldc + col] = (_Float16)v;
          } else if (EPI == EPI_TANH) {
            const float* bi = p1 + (long)z * pStrideZ;
            v = tanhf(v + bi[col]);
            outH[(long)row * ldc + col] = (_Float16)v;
          } else if (EPI == EPI_BIASF32) {
            const float* bi = p1 + (long)z * pStrideZ;
            outF[(long)row * ldc + col] = v + bi[col];
          } else {
            outH[(long)row * ldc + col] = (_Float16)v;
          }
        } else if (col < nPad) {
          if (EPI == EPI_BIASF32) outF[(long)row * ldc + col] = 0.f;
          else                    outH[(long)row * ldc + col] = (_Float16)0.f;
        }
      }
    }
  }
}

// ---------------------------------------------------------------------------
// Weight prep
// ---------------------------------------------------------------------------
// OIHW f32 -> [Kpad][CO] f16 with k = (kh*3+kw)*CI + ci (matches im2col order)
__global__ void pack_wk(const float* __restrict__ W, _Float16* __restrict__ Wk,
                        int CI, int CO, int Kpad) {
  long idx = (long)blockIdx.x * blockDim.x + threadIdx.x;
  if (idx >= (long)Kpad * CO) return;
  int co = (int)(idx % CO);
  int k  = (int)(idx / CO);
  _Float16 v = (_Float16)0.f;
  if (k < 9 * CI) {
    int ci = k % CI, s = k / CI, kh = s / 3, kw = s % 3;
    v = (_Float16)W[(((long)co * CI + ci) * 3 + kh) * 3 + kw];
  }
  Wk[idx] = v;
}

__global__ void bn_params(const float* __restrict__ b, const float* __restrict__ g,
                          const float* __restrict__ be, const float* __restrict__ m,
                          const float* __restrict__ v, float* __restrict__ alpha,
                          float* __restrict__ beta, int C) {
  int i = blockIdx.x * blockDim.x + threadIdx.x;
  if (i >= C) return;
  float a = g[i] * rsqrtf(v[i] + 1e-5f);
  alpha[i] = a;
  beta[i]  = be[i] + (b[i] - m[i]) * a;
}

// Wa f32 [25,336,600] -> f16 [25,352,600] (zero K-pad rows)
__global__ void cvt_wa(const float* __restrict__ Wa, _Float16* __restrict__ out) {
  long idx = (long)blockIdx.x * blockDim.x + threadIdx.x;
  if (idx >= 25L * 352 * 600) return;
  int h = (int)(idx % 600);
  long t = idx / 600;
  int i = (int)(t % 352);
  int n = (int)(t / 352);
  out[idx] = (i < 336) ? (_Float16)Wa[((long)n * 336 + i) * 600 + h] : (_Float16)0.f;
}

// Wb f32 [25,600,10] -> f16 [25,608,16] (zero pads)
__global__ void cvt_wb(const float* __restrict__ Wb, _Float16* __restrict__ out) {
  long idx = (long)blockIdx.x * blockDim.x + threadIdx.x;
  if (idx >= 25L * 608 * 16) return;
  int c = (int)(idx % 16);
  long t = idx / 16;
  int i = (int)(t % 608);
  int n = (int)(t / 608);
  out[idx] = (i < 600 && c < 10) ? (_Float16)Wb[((long)n * 600 + i) * 10 + c] : (_Float16)0.f;
}

__global__ void pad_bb(const float* __restrict__ bb, float* __restrict__ out) {
  int idx = blockIdx.x * blockDim.x + threadIdx.x;
  if (idx >= 25 * 16) return;
  int c = idx % 16, n = idx / 16;
  out[idx] = (c < 10) ? bb[n * 10 + c] : 0.f;
}

// ---------------------------------------------------------------------------
// im2col / pooling / pointwise
// ---------------------------------------------------------------------------
// conv1: x f32 NCHW [B,3,20,20] -> patches [B*400][32], k = (kh*3+kw)*3 + ci
__global__ void im2col_conv1(const float* __restrict__ x, _Float16* __restrict__ out, long Bn) {
  long idx = (long)blockIdx.x * blockDim.x + threadIdx.x;
  if (idx >= Bn * 400 * 32) return;
  int k = (int)(idx % 32);
  long p = idx / 32;
  int px = (int)(p % 20);
  long t = p / 20;
  int py = (int)(t % 20);
  long b = t / 20;
  _Float16 v = (_Float16)0.f;
  if (k < 27) {
    int s = k / 3, ci = k % 3, kh = s / 3, kw = s % 3;
    int iy = py + kh - 1, ix = px + kw - 1;
    if (iy >= 0 && iy < 20 && ix >= 0 && ix < 20)
      v = (_Float16)x[((b * 3 + ci) * 20 + iy) * 20 + ix];
  }
  out[idx] = v;
}

// NHWC f16 -> patches [B*H*W][9*C], k = (kh*3+kw)*C + c ; one thread = 8 channels
__global__ void im2col_nhwc(const _Float16* __restrict__ in, _Float16* __restrict__ out,
                            int Bn, int H, int W, int C) {
  long idx = (long)blockIdx.x * blockDim.x + threadIdx.x;
  int c8n = C / 8;
  long total = (long)Bn * H * W * 9 * c8n;
  if (idx >= total) return;
  int c8 = (int)(idx % c8n);
  long t = idx / c8n;
  int s = (int)(t % 9);
  long p = t / 9;
  int x = (int)(p % W);
  long t2 = p / W;
  int y = (int)(t2 % H);
  long b = t2 / H;
  int kh = s / 3, kw = s % 3;
  int iy = y + kh - 1, ix = x + kw - 1;
  h8 v = {};
  if (iy >= 0 && iy < H && ix >= 0 && ix < W)
    v = *(const h8*)&in[(((long)b * H + iy) * W + ix) * C + c8 * 8];
  *(h8*)&out[p * (9L * C) + (long)s * C + c8 * 8] = v;
}

// MaxPool2d(3, stride 2, pad 1), NHWC f16
__global__ void maxpool_nhwc(const _Float16* __restrict__ in, _Float16* __restrict__ out,
                             int Bn, int H, int W, int C) {
  int OH = H / 2, OW = W / 2;
  long idx = (long)blockIdx.x * blockDim.x + threadIdx.x;
  long total = (long)Bn * OH * OW * C;
  if (idx >= total) return;
  int c = (int)(idx % C);
  long t = idx / C;
  int ox = (int)(t % OW);
  t /= OW;
  int oy = (int)(t % OH);
  long b = t / OH;
  float mx = -INFINITY;
#pragma unroll
  for (int kh = 0; kh < 3; ++kh) {
#pragma unroll
    for (int kw = 0; kw < 3; ++kw) {
      int iy = oy * 2 + kh - 1, ix = ox * 2 + kw - 1;
      if (iy >= 0 && iy < H && ix >= 0 && ix < W) {
        float v = (float)in[(((long)b * H + iy) * W + ix) * C + c];
        mx = fmaxf(mx, v);
      }
    }
  }
  out[idx] = (_Float16)mx;
}

// softmax over the first 10 of 16-strided logit rows
__global__ void softmax10(const float* __restrict__ lg, float* __restrict__ out, long rows) {
  long r = (long)blockIdx.x * blockDim.x + threadIdx.x;
  if (r >= rows) return;
  const float* p = lg + r * 16;
  float mx = p[0];
#pragma unroll
  for (int i = 1; i < 10; ++i) mx = fmaxf(mx, p[i]);
  float e[10], s = 0.f;
#pragma unroll
  for (int i = 0; i < 10; ++i) { e[i] = expf(p[i] - mx); s += e[i]; }
  float inv = 1.f / s;
  float* q = out + r * 10;
#pragma unroll
  for (int i = 0; i < 10; ++i) q[i] = e[i] * inv;
}

__device__ inline void build_nei(int i, int* nn) {
  const int w = 5, size = 25;
  int c = 0;
  if (i - w >= 0) nn[c++] = i - w;
  if (i % w != 0) nn[c++] = i - 1;
  if ((i + 1) % w != 0) nn[c++] = i + 1;
  if (i + w < size) nn[c++] = i + w;
  if (i - w - 1 >= 0 && i % w != 0) nn[c++] = i - w - 1;
  if (i - w + 1 >= 0 && (i + 1) % w != 0) nn[c++] = i - w + 1;
  if (i + w - 1 < size && i % w != 0) nn[c++] = i + w - 1;
  if (i + w + 1 < size && (i + 1) % w != 0) nn[c++] = i + w + 1;
  while (c < 8) nn[c++] = -1;
}

// stage-2 input: s2in[n][b][0..79]=neighbor preds, [80..335]=feats, [336..351]=0
__global__ void build_s2in(const float* __restrict__ preds, const _Float16* __restrict__ feats,
                           _Float16* __restrict__ out, long Bn) {
  long idx = (long)blockIdx.x * blockDim.x + threadIdx.x;
  if (idx >= 25L * Bn * 352) return;
  int j = (int)(idx % 352);
  long t = idx / 352;
  long b = t % Bn;
  int n = (int)(t / Bn);
  _Float16 v = (_Float16)0.f;
  if (j < 80) {
    int jn = j / 10, c = j % 10;
    int nn[8];
    build_nei(n, nn);
    int nb = nn[jn];
    if (nb >= 0) v = (_Float16)preds[((long)nb * Bn + b) * 10 + c];
  } else if (j < 336) {
    v = feats[(b * 25 + n) * 256 + (j - 80)];
  }
  out[idx] = v;
}

__global__ void mean_nodes(const float* __restrict__ second, float* __restrict__ out, long Bn) {
  long idx = (long)blockIdx.x * blockDim.x + threadIdx.x;
  if (idx >= Bn * 10) return;
  float s = 0.f;
#pragma unroll
  for (int n = 0; n < 25; ++n) s += second[(long)n * Bn * 10 + idx];
  out[idx] = s * (1.f / 25.f);
}

static inline unsigned int nblk(long total) { return (unsigned int)((total + TB - 1) / TB); }

// ---------------------------------------------------------------------------
extern "C" void kernel_launch(void* const* d_in, const int* in_sizes, int n_in,
                              void* d_out, int out_size, void* d_ws, size_t ws_size,
                              hipStream_t stream) {
  (void)n_in; (void)out_size; (void)ws_size;
  const float* x = (const float*)d_in[0];
  const float *Wc[4], *bc[4], *gc[4], *bec[4], *mc[4], *vc[4];
  for (int l = 0; l < 4; ++l) {
    Wc[l]  = (const float*)d_in[1 + 6 * l + 0];
    bc[l]  = (const float*)d_in[1 + 6 * l + 1];
    gc[l]  = (const float*)d_in[1 + 6 * l + 2];
    bec[l] = (const float*)d_in[1 + 6 * l + 3];
    mc[l]  = (const float*)d_in[1 + 6 * l + 4];
    vc[l]  = (const float*)d_in[1 + 6 * l + 5];
  }
  const float* Wa = (const float*)d_in[25];
  const float* ba = (const float*)d_in[26];
  const float* Wb = (const float*)d_in[27];
  const float* bb = (const float*)d_in[28];

  const long Bn = (long)in_sizes[0] / (3 * 20 * 20);  // 8192

  // ---- workspace layout (bump allocate, lifetime reuse) ----
  char* ws = (char*)d_ws;
  size_t off = 0;
  auto alloc = [&](size_t bytes) -> char* {
    char* p = ws + off;
    off += (bytes + 255) & ~(size_t)255;
    return p;
  };
  const size_t patchBytes = (size_t)Bn * 25 * 2304 * 2;  // max im2col (conv2 == conv4)
  char* patch = alloc(patchBytes);
  // act1 lives in the tail of the patch region (conv1 patches use only Bn*400*32*2 bytes)
  _Float16* act1 = (_Float16*)(patch + (size_t)Bn * 400 * 32 * 2);
  const size_t hBytes = (size_t)25 * Bn * 608 * 2;
  char* regH = alloc(hBytes + (size_t)25 * Bn * 16 * 4 + 256);  // h + logits
  _Float16* hbuf = (_Float16*)regH;
  float* logits = (float*)(regH + ((hBytes + 255) & ~(size_t)255));
  _Float16* pool1 = (_Float16*)alloc((size_t)Bn * 100 * 64 * 2);
  char* regB = alloc((size_t)Bn * 100 * 128 * 2);  // act2, later s2in
  _Float16* act2 = (_Float16*)regB;
  _Float16* s2in = (_Float16*)regB;
  _Float16* pool2 = (_Float16*)alloc((size_t)Bn * 25 * 128 * 2);
  char* regC = alloc((size_t)Bn * 25 * 256 * 2);  // act3, later preds
  _Float16* act3 = (_Float16*)regC;
  float* preds = (float*)regC;
  _Float16* feats = (_Float16*)alloc((size_t)Bn * 25 * 256 * 2);
  const int CO[4] = {64, 128, 256, 256};
  const int CI[4] = {3, 64, 128, 256};
  const int Kc[4] = {32, 576, 1152, 2304};
  _Float16* wk[4];
  float *alp[4], *bet[4];
  for (int l = 0; l < 4; ++l) wk[l] = (_Float16*)alloc((size_t)Kc[l] * CO[l] * 2);
  for (int l = 0; l < 4; ++l) { alp[l] = (float*)alloc(CO[l] * 4); bet[l] = (float*)alloc(CO[l] * 4); }
  _Float16* waf = (_Float16*)alloc((size_t)25 * 352 * 600 * 2);
  _Float16* wbf = (_Float16*)alloc((size_t)25 * 608 * 16 * 2);
  float* bbp = (float*)alloc(25 * 16 * 4);

  // ---- weight prep ----
  for (int l = 0; l < 4; ++l) {
    pack_wk<<<nblk((long)Kc[l] * CO[l]), TB, 0, stream>>>(Wc[l], wk[l], CI[l], CO[l], Kc[l]);
    bn_params<<<nblk(CO[l]), TB, 0, stream>>>(bc[l], gc[l], bec[l], mc[l], vc[l], alp[l], bet[l], CO[l]);
  }
  cvt_wa<<<nblk(25L * 352 * 600), TB, 0, stream>>>(Wa, waf);
  cvt_wb<<<nblk(25L * 608 * 16), TB, 0, stream>>>(Wb, wbf);
  pad_bb<<<nblk(25 * 16), TB, 0, stream>>>(bb, bbp);

  // ---- conv1: [B,3,20,20] -> act1 NHWC [B,20,20,64] ----
  im2col_conv1<<<nblk(Bn * 400 * 32), TB, 0, stream>>>(x, (_Float16*)patch, Bn);
  gemm_wmma<EPI_BNRELU><<<dim3(1, (unsigned)(Bn * 400 / TBM), 1), TB, GEMM_LDS_BYTES, stream>>>(
      (const _Float16*)patch, 32, 0, wk[0], 64, 0, act1, 64, 0,
      (int)(Bn * 400), 64, 32, 64, alp[0], bet[0], 0);
  maxpool_nhwc<<<nblk(Bn * 100 * 64), TB, 0, stream>>>(act1, pool1, (int)Bn, 20, 20, 64);

  // ---- conv2 -> act2 [B,10,10,128] ----
  im2col_nhwc<<<nblk(Bn * 100 * 9 * 8), TB, 0, stream>>>(pool1, (_Float16*)patch, (int)Bn, 10, 10, 64);
  gemm_wmma<EPI_BNRELU><<<dim3(2, (unsigned)(Bn * 100 / TBM), 1), TB, GEMM_LDS_BYTES, stream>>>(
      (const _Float16*)patch, 576, 0, wk[1], 128, 0, act2, 128, 0,
      (int)(Bn * 100), 128, 576, 128, alp[1], bet[1], 0);
  maxpool_nhwc<<<nblk(Bn * 25 * 128), TB, 0, stream>>>(act2, pool2, (int)Bn, 10, 10, 128);

  // ---- conv3 -> act3 [B,5,5,256] ----
  im2col_nhwc<<<nblk(Bn * 25 * 9 * 16), TB, 0, stream>>>(pool2, (_Float16*)patch, (int)Bn, 5, 5, 128);
  gemm_wmma<EPI_BNRELU><<<dim3(4, (unsigned)(Bn * 25 / TBM), 1), TB, GEMM_LDS_BYTES, stream>>>(
      (const _Float16*)patch, 1152, 0, wk[2], 256, 0, act3, 256, 0,
      (int)(Bn * 25), 256, 1152, 256, alp[2], bet[2], 0);

  // ---- conv4 -> feats [B,25,256] (NHWC == feats[node,b,c] transposed view) ----
  im2col_nhwc<<<nblk(Bn * 25 * 9 * 32), TB, 0, stream>>>(act3, (_Float16*)patch, (int)Bn, 5, 5, 256);
  gemm_wmma<EPI_BNRELU><<<dim3(4, (unsigned)(Bn * 25 / TBM), 1), TB, GEMM_LDS_BYTES, stream>>>(
      (const _Float16*)patch, 2304, 0, wk[3], 256, 0, feats, 256, 0,
      (int)(Bn * 25), 256, 2304, 256, alp[3], bet[3], 0);

  float* second = (float*)d_out + Bn * 10;

  // ---- stage 1 MLP (first 80 inputs are zero -> K=256, Wa rows 80..335) ----
  gemm_wmma<EPI_TANH><<<dim3(10, (unsigned)(Bn / TBM), 25), TB, GEMM_LDS_BYTES, stream>>>(
      feats, 25 * 256, 256, waf + 80 * 600, 600, (long)352 * 600,
      hbuf, 608, (long)Bn * 608, (int)Bn, 600, 256, 608, nullptr, ba, 600);
  gemm_wmma<EPI_BIASF32><<<dim3(1, (unsigned)(Bn / TBM), 25), TB, GEMM_LDS_BYTES, stream>>>(
      hbuf, 608, (long)Bn * 608, wbf, 16, (long)608 * 16,
      logits, 16, (long)Bn * 16, (int)Bn, 16, 608, 16, nullptr, bbp, 16);
  softmax10<<<nblk(25 * Bn), TB, 0, stream>>>(logits, preds, 25 * Bn);

  // ---- stage 2 ----
  build_s2in<<<nblk(25L * Bn * 352), TB, 0, stream>>>(preds, feats, s2in, Bn);
  gemm_wmma<EPI_TANH><<<dim3(10, (unsigned)(Bn / TBM), 25), TB, GEMM_LDS_BYTES, stream>>>(
      s2in, 352, (long)Bn * 352, waf, 600, (long)352 * 600,
      hbuf, 608, (long)Bn * 608, (int)Bn, 600, 352, 608, nullptr, ba, 600);
  gemm_wmma<EPI_BIASF32><<<dim3(1, (unsigned)(Bn / TBM), 25), TB, GEMM_LDS_BYTES, stream>>>(
      hbuf, 608, (long)Bn * 608, wbf, 16, (long)608 * 16,
      logits, 16, (long)Bn * 16, (int)Bn, 16, 608, 16, nullptr, bbp, 16);
  softmax10<<<nblk(25 * Bn), TB, 0, stream>>>(logits, second, 25 * Bn);

  // ---- mean over nodes ----
  mean_nodes<<<nblk(Bn * 10), TB, 0, stream>>>(second, (float*)d_out, Bn);
}
